// MoRAttention_22557168239273
// MI455X (gfx1250) — compile-verified, matrix operand order checked by reading, not run on hardware
//
#include <hip/hip_runtime.h>
#include <stdint.h>
#include <stddef.h>

// ---- problem constants (match reference) ----
#define B_   2
#define S_   2048
#define H_   2048
#define NH_  16
#define NKV_ 4
#define HD_  128
#define GRP_ (NH_ / NKV_)
#define M_   (B_ * S_)

typedef __attribute__((ext_vector_type(16))) __bf16 v16bf;
typedef __attribute__((ext_vector_type(8)))  __bf16 bf16x8;
typedef __attribute__((ext_vector_type(8)))  float  v8f;
typedef __attribute__((ext_vector_type(4)))  int    i32x4;
typedef __attribute__((ext_vector_type(8)))  int    i32x8;

typedef __attribute__((address_space(3))) const void lds_cv_t;

union AFrag { v16bf v; bf16x8 h[2]; };

__device__ __forceinline__ v8f zero8() {
  v8f z;
#pragma unroll
  for (int i = 0; i < 8; ++i) z[i] = 0.0f;
  return z;
}

// LDS byte offset of a generic pointer into shared memory (addrspacecast -> as3).
__device__ __forceinline__ uint32_t lds_off_u32(const void* p) {
  return (uint32_t)(uintptr_t)(lds_cv_t*)p;
}

// ---- CDNA5 per-lane async global->LDS (ASYNCcnt), via inline asm ----
// VDST = per-lane LDS byte address, VADDR = 64-bit global address (GV mode).
__device__ __forceinline__ void async_b128(const void* g, void* l) {
  asm volatile("global_load_async_to_lds_b128 %0, %1, off"
               ::"v"(lds_off_u32(l)), "v"((uint64_t)(uintptr_t)g)
               : "memory");
}
#define WAIT_ASYNC(n) asm volatile("s_wait_asynccnt " #n ::: "memory")

// ---- CDNA5 Tensor Data Mover: 2D tile (2-byte elements) -> LDS ----
// D# per CDNA5 ISA: group0 {count=1, lds_addr, global_addr,
// type=2}; group1 {data_size=1(2B), tensor_dim0/1, tile_dim0/1, dim0_stride}.
__device__ __forceinline__ void tdm_load_tile_2d_bf16(
    const __bf16* gsrc, uint32_t lds_byte_addr,
    uint32_t tile_w, uint32_t tile_h, uint32_t row_stride_elems) {
  uint64_t ga = (uint64_t)(uintptr_t)gsrc;
  i32x4 g0;
  g0[0] = 1;                               // count=1 (valid user descriptor)
  g0[1] = (int)lds_byte_addr;              // LDS destination (bytes)
  g0[2] = (int)(uint32_t)ga;               // global_addr[31:0]
  g0[3] = (int)((((uint32_t)(ga >> 32)) & 0x01FFFFFFu) | (2u << 30)); // addr[56:32], type=2
  i32x8 g1;
  g1[0] = (int)(1u << 16);                                  // data_size=1 -> 2 bytes
  g1[1] = (int)((tile_w & 0xFFFFu) << 16);                  // tensor_dim0[15:0]
  g1[2] = (int)((tile_w >> 16) | ((tile_h & 0xFFFFu) << 16)); // dim0[31:16] | dim1[15:0]
  g1[3] = (int)((tile_h >> 16) | (tile_w << 16));           // dim1[31:16] | tile_dim0
  g1[4] = (int)(tile_h & 0xFFFFu);                          // tile_dim1 (tile_dim2=0)
  g1[5] = (int)row_stride_elems;                            // tensor_dim0_stride[31:0]
  g1[6] = 0;                                                // stride[47:32] | dim1_stride lo
  g1[7] = 0;
  // 2D tensor: VADDR2/VADDR3 omitted (both-NULL form).
  asm volatile("tensor_load_to_lds %0, %1" ::"s"(g0), "s"(g1) : "memory");
}

// ------------------------------------------------------------------
// f32 -> bf16 elementwise convert
// ------------------------------------------------------------------
__global__ __launch_bounds__(256) void cvt_bf16_kernel(const float* __restrict__ in,
                                                       __bf16* __restrict__ out,
                                                       int n) {
  int i = blockIdx.x * 256 + threadIdx.x;
  if (i < n) out[i] = (__bf16)in[i];
}

// ------------------------------------------------------------------
// C[M,N](f32) = A[M,K](bf16) @ W[N,K]^T(bf16)
// BM=128 x BN=64 tile, KC=32, 8 waves/WG, double-buffered LDS.
// A tile: TDM (tensor_load_to_lds) issued by wave 0, TENSORcnt-waited.
// B tile: per-lane async global->LDS (ASYNCcnt), depth-1 pipelined.
// ------------------------------------------------------------------
#define BM 128
#define BN 64
#define KC 32

__global__ __launch_bounds__(256) void gemm_bf16_f32_kernel(
    const __bf16* __restrict__ A, const __bf16* __restrict__ W,
    float* __restrict__ C, int M, int N, int K) {
  __shared__ __bf16 As[2][BM * KC];  // 2 x 8 KB
  __shared__ __bf16 Bs[2][BN * KC];  // 2 x 4 KB

  const int tid  = threadIdx.x;
  const int lane = tid & 31;
  const int wave = tid >> 5;   // 0..7 -> 16 output rows each
  const int g    = lane >> 4;  // lane half
  const int ln   = lane & 15;
  const int m0 = blockIdx.y * BM;
  const int n0 = blockIdx.x * BN;

  v8f acc[4];
#pragma unroll
  for (int j = 0; j < 4; ++j) acc[j] = zero8();

  const int brow = tid >> 2, bch = (tid & 3) * 8;  // B tile: one b128 per thread

  auto issue_tile = [&](int buf, int k0) {
    if (wave == 0) {  // one TDM per workgroup tile (EXEC ignored, wave-level op)
      tdm_load_tile_2d_bf16(A + (size_t)m0 * K + k0,
                            lds_off_u32(&As[buf][0]),
                            KC, BM, (uint32_t)K);
    }
    async_b128(W + (size_t)(n0 + brow) * K + k0 + bch, &Bs[buf][brow * KC + bch]);
  };

  const int nk = K / KC;
  issue_tile(0, 0);
  for (int kb = 0; kb < nk; ++kb) {
    const int buf = kb & 1;
    if (kb + 1 < nk) {
      issue_tile(buf ^ 1, (kb + 1) * KC);  // prefetch next tile
      if (wave == 0) __builtin_amdgcn_s_wait_tensorcnt(1);
      WAIT_ASYNC(1);                       // this tile's data landed
    } else {
      if (wave == 0) __builtin_amdgcn_s_wait_tensorcnt(0);
      WAIT_ASYNC(0);
    }
    __syncthreads();  // publish LDS tile to all waves

    // A fragment: lane holds row (wave*16+ln); K chunks {8g..8g+7, 16+8g..}
    AFrag a;
    {
      const __bf16* base = &As[buf][(wave * 16 + ln) * KC];
      a.h[0] = *(const bf16x8*)(base + g * 8);
      a.h[1] = *(const bf16x8*)(base + 16 + g * 8);
    }
#pragma unroll
    for (int j = 0; j < 4; ++j) {
      AFrag bfr;  // B fragment: lane holds column (j*16+ln)
      const __bf16* base = &Bs[buf][(j * 16 + ln) * KC];
      bfr.h[0] = *(const bf16x8*)(base + g * 8);
      bfr.h[1] = *(const bf16x8*)(base + 16 + g * 8);
      acc[j] = __builtin_amdgcn_wmma_f32_16x16x32_bf16(
          false, a.v, false, bfr.v, (short)0, acc[j], false, false);
    }
    __syncthreads();  // all reads done before this buffer is refilled
  }

#pragma unroll
  for (int j = 0; j < 4; ++j)
#pragma unroll
    for (int r = 0; r < 8; ++r)
      C[(size_t)(m0 + wave * 16 + g * 8 + r) * N + (n0 + j * 16 + ln)] = acc[j][r];
}

// ------------------------------------------------------------------
// RoPE (optional) + f32->bf16 pack + layout change.
// transpose_out==0 -> [B, nh, S, HD] (Q, K); ==1 -> [B, nh, HD, S] (V)
// ------------------------------------------------------------------
__global__ __launch_bounds__(256) void rope_cvt_kernel(const float* __restrict__ Xf,
                                                       __bf16* __restrict__ Out,
                                                       int nh, int apply_rope,
                                                       int transpose_out) {
  size_t idx = (size_t)blockIdx.x * 256 + threadIdx.x;
  int d = (int)(idx & (HD_ - 1));
  size_t t = idx >> 7;  // (b*S+s)*nh + hh
  int hh = (int)(t % nh);
  size_t t2 = t / nh;   // b*S + s
  int s = (int)(t2 & (S_ - 1));
  int b = (int)(t2 >> 11);

  const float* row = Xf + t2 * (size_t)(nh * HD_) + hh * HD_;
  float x = row[d];
  float o = x;
  if (apply_rope) {
    int j = d & 63;
    float inv = __powf(10000.0f, -(float)(2 * j) * (1.0f / 128.0f));
    float ang = (float)s * inv;
    float sn, cs;
    __sincosf(ang, &sn, &cs);
    float xr = (d < 64) ? -row[d + 64] : row[d - 64];  // rotate_half
    o = x * cs + xr * sn;
  }
  size_t oidx = transpose_out
      ? ((((size_t)(b * nh + hh)) * HD_ + d) * S_ + s)
      : ((((size_t)(b * nh + hh)) * S_ + s) * HD_ + d);
  Out[oidx] = (__bf16)o;
}

// ------------------------------------------------------------------
// Flash attention: 1 wave per (b, h, 16-row q-tile). Key blocks of 32.
// ------------------------------------------------------------------
__global__ __launch_bounds__(32) void flash_attn_kernel(
    const __bf16* __restrict__ Q,    // [B,NH,S,HD]
    const __bf16* __restrict__ Kt,   // [B,NKV,S,HD]
    const __bf16* __restrict__ Vt,   // [B,NKV,HD,S]
    const float* __restrict__ mask,  // [S,S]
    __bf16* __restrict__ Ctx)        // [B*S, NH*HD]
{
  __shared__ __bf16 Pl[16 * 32];  // P bounce buffer (C-layout -> A-frag layout)

  const int lane = threadIdx.x;
  const int g = lane >> 4;
  const int ln = lane & 15;
  const int qt = blockIdx.x;
  const int h  = blockIdx.y;
  const int b  = blockIdx.z;
  const int kv = h / GRP_;

  const __bf16* Qbase = Q + (((size_t)(b * NH_ + h)) * S_ + qt * 16) * HD_;
  const __bf16* Kbase = Kt + ((size_t)(b * NKV_ + kv)) * S_ * HD_;
  const __bf16* Vbase = Vt + ((size_t)(b * NKV_ + kv)) * (size_t)HD_ * S_;

  AFrag qf[4];
#pragma unroll
  for (int dc = 0; dc < 4; ++dc) {
    const __bf16* r = Qbase + (size_t)ln * HD_ + dc * 32;
    qf[dc].h[0] = *(const bf16x8*)(r + g * 8);
    qf[dc].h[1] = *(const bf16x8*)(r + 16 + g * 8);
  }

  float m_i[8], l_i[8];
  v8f acc[8];
#pragma unroll
  for (int r = 0; r < 8; ++r) { m_i[r] = -1e30f; l_i[r] = 0.0f; }
#pragma unroll
  for (int t = 0; t < 8; ++t) acc[t] = zero8();

  const float scale = 0.08838834764831845f;  // 1/sqrt(128)

  for (int kb = 0; kb < S_; kb += 32) {
    v8f sc[2];
#pragma unroll
    for (int j = 0; j < 2; ++j) {
      v8f s = zero8();
#pragma unroll
      for (int dc = 0; dc < 4; ++dc) {
        AFrag kf;
        const __bf16* r = Kbase + (size_t)(kb + j * 16 + ln) * HD_ + dc * 32;
        kf.h[0] = *(const bf16x8*)(r + g * 8);
        kf.h[1] = *(const bf16x8*)(r + 16 + g * 8);
        s = __builtin_amdgcn_wmma_f32_16x16x32_bf16(
            false, qf[dc].v, false, kf.v, (short)0, s, false, false);
      }
      sc[j] = s;
    }

    float rowmax[8];
#pragma unroll
    for (int r = 0; r < 8; ++r) {
      const float* mrow = mask + (size_t)(qt * 16 + g * 8 + r) * S_ + kb;
      float v0 = sc[0][r] * scale + mrow[ln];
      float v1 = sc[1][r] * scale + mrow[16 + ln];
      sc[0][r] = v0; sc[1][r] = v1;
      rowmax[r] = fmaxf(v0, v1);
    }
#pragma unroll
    for (int off = 8; off > 0; off >>= 1)
#pragma unroll
      for (int r = 0; r < 8; ++r)
        rowmax[r] = fmaxf(rowmax[r], __shfl_xor(rowmax[r], off, 16));

    float corr[8], rowsum[8];
#pragma unroll
    for (int r = 0; r < 8; ++r) {
      float mnew = fmaxf(m_i[r], rowmax[r]);
      corr[r] = __expf(m_i[r] - mnew);
      m_i[r] = mnew;
      float p0 = __expf(sc[0][r] - mnew);
      float p1 = __expf(sc[1][r] - mnew);
      sc[0][r] = p0; sc[1][r] = p1;
      rowsum[r] = p0 + p1;
    }
#pragma unroll
    for (int off = 8; off > 0; off >>= 1)
#pragma unroll
      for (int r = 0; r < 8; ++r)
        rowsum[r] += __shfl_xor(rowsum[r], off, 16);
#pragma unroll
    for (int r = 0; r < 8; ++r) l_i[r] = l_i[r] * corr[r] + rowsum[r];
#pragma unroll
    for (int t = 0; t < 8; ++t)
#pragma unroll
      for (int r = 0; r < 8; ++r) acc[t][r] *= corr[r];

    // P: C-layout -> A-frag layout via LDS (DS ops are wave-in-order)
#pragma unroll
    for (int r = 0; r < 8; ++r) {
      Pl[(g * 8 + r) * 32 + ln]      = (__bf16)sc[0][r];
      Pl[(g * 8 + r) * 32 + 16 + ln] = (__bf16)sc[1][r];
    }
    __builtin_amdgcn_wave_barrier();
    AFrag pf;
    {
      const __bf16* base = &Pl[ln * 32];
      pf.h[0] = *(const bf16x8*)(base + g * 8);
      pf.h[1] = *(const bf16x8*)(base + 16 + g * 8);
    }
    __builtin_amdgcn_wave_barrier();

#pragma unroll
    for (int t = 0; t < 8; ++t) {
      AFrag vf;  // B-frag from transposed V: contiguous over keys
      const __bf16* col = Vbase + (size_t)(t * 16 + ln) * S_ + kb;
      vf.h[0] = *(const bf16x8*)(col + g * 8);
      vf.h[1] = *(const bf16x8*)(col + 16 + g * 8);
      acc[t] = __builtin_amdgcn_wmma_f32_16x16x32_bf16(
          false, pf.v, false, vf.v, (short)0, acc[t], false, false);
    }
  }

#pragma unroll
  for (int t = 0; t < 8; ++t) {
#pragma unroll
    for (int r = 0; r < 8; ++r) {
      float o = acc[t][r] / l_i[r];
      size_t row = (size_t)(b * S_ + qt * 16 + g * 8 + r);
      Ctx[row * (NH_ * HD_) + h * HD_ + t * 16 + ln] = (__bf16)o;
    }
  }
}

// ------------------------------------------------------------------
// Host launcher
// ------------------------------------------------------------------
extern "C" void kernel_launch(void* const* d_in, const int* in_sizes, int n_in,
                              void* d_out, int out_size, void* d_ws, size_t ws_size,
                              hipStream_t stream) {
  (void)in_sizes; (void)n_in; (void)out_size; (void)ws_size;
  const float* hs   = (const float*)d_in[0];
  const float* mask = (const float*)d_in[1];
  const float* Wq   = (const float*)d_in[2];
  const float* Wk   = (const float*)d_in[3];
  const float* Wv   = (const float*)d_in[4];
  const float* Wo   = (const float*)d_in[5];
  float* out = (float*)d_out;

  char* p = (char*)d_ws;
  auto take = [&](size_t bytes) {
    char* r = p;
    p += (bytes + 255) & ~(size_t)255;
    return r;
  };

  __bf16* Xb  = (__bf16*)take((size_t)M_ * H_ * 2);
  __bf16* Wqb = (__bf16*)take((size_t)H_ * H_ * 2);
  __bf16* Wkb = (__bf16*)take((size_t)(NKV_ * HD_) * H_ * 2);
  __bf16* Wvb = (__bf16*)take((size_t)(NKV_ * HD_) * H_ * 2);
  __bf16* Wob = (__bf16*)take((size_t)H_ * H_ * 2);
  float*  Qf  = (float*)take((size_t)M_ * H_ * 4);
  float*  Kf  = (float*)take((size_t)M_ * (NKV_ * HD_) * 4);
  float*  Vf  = (float*)take((size_t)M_ * (NKV_ * HD_) * 4);
  __bf16* Qb  = (__bf16*)take((size_t)M_ * H_ * 2);
  __bf16* Kb  = (__bf16*)take((size_t)M_ * (NKV_ * HD_) * 2);
  __bf16* Vtb = (__bf16*)take((size_t)M_ * (NKV_ * HD_) * 2);
  __bf16* Ctx = (__bf16*)Qf;  // alias: Qf is dead once Qb is built

  auto cvt = [&](const float* src, __bf16* dst, int n) {
    cvt_bf16_kernel<<<(n + 255) / 256, 256, 0, stream>>>(src, dst, n);
  };
  cvt(hs, Xb, M_ * H_);
  cvt(Wq, Wqb, H_ * H_);
  cvt(Wk, Wkb, NKV_ * HD_ * H_);
  cvt(Wv, Wvb, NKV_ * HD_ * H_);
  cvt(Wo, Wob, H_ * H_);

  dim3 blk(256);
  gemm_bf16_f32_kernel<<<dim3(H_ / BN, M_ / BM), blk, 0, stream>>>(
      Xb, Wqb, Qf, M_, H_, H_);
  gemm_bf16_f32_kernel<<<dim3((NKV_ * HD_) / BN, M_ / BM), blk, 0, stream>>>(
      Xb, Wkb, Kf, M_, NKV_ * HD_, H_);
  gemm_bf16_f32_kernel<<<dim3((NKV_ * HD_) / BN, M_ / BM), blk, 0, stream>>>(
      Xb, Wvb, Vf, M_, NKV_ * HD_, H_);

  rope_cvt_kernel<<<(int)(((size_t)M_ * NH_ * HD_) / 256), 256, 0, stream>>>(
      Qf, Qb, NH_, 1, 0);
  rope_cvt_kernel<<<(int)(((size_t)M_ * NKV_ * HD_) / 256), 256, 0, stream>>>(
      Kf, Kb, NKV_, 1, 0);
  rope_cvt_kernel<<<(int)(((size_t)M_ * NKV_ * HD_) / 256), 256, 0, stream>>>(
      Vf, Vtb, NKV_, 0, 1);

  flash_attn_kernel<<<dim3(S_ / 16, NH_, B_), dim3(32), 0, stream>>>(
      Qb, Kb, Vtb, mask, Ctx);

  gemm_bf16_f32_kernel<<<dim3(H_ / BN, M_ / BM), blk, 0, stream>>>(
      Ctx, Wob, out, M_, H_, H_);
}
